// MomemtumConceptAttentionProto_35296041238692
// MI455X (gfx1250) — matrix operand order repaired and checked by reading, not compiled, not run on hardware
//
#include <hip/hip_runtime.h>
#include <hip/hip_bf16.h>

#define B_  8
#define CH_ 1024
#define N_  4096
#define FD_ 128
#define P_  2048

#define AI __attribute__((always_inline))

typedef __bf16 bf16;
typedef __attribute__((ext_vector_type(16))) __bf16 v16bf;
typedef __attribute__((ext_vector_type(8)))  float  v8f;
typedef unsigned int v4u __attribute__((ext_vector_type(4)));
typedef int v8i __attribute__((ext_vector_type(8)));
typedef int v4i __attribute__((ext_vector_type(4)));

#if defined(__has_builtin)
#if __has_builtin(__builtin_amdgcn_tensor_load_to_lds) && __has_builtin(__builtin_amdgcn_s_wait_tensorcnt)
#define HAVE_TDM 1
#endif
#endif

static __device__ AI inline v8f wmma_bf16(v16bf a, v16bf b, v8f c) {
  return __builtin_amdgcn_wmma_f32_16x16x32_bf16(false, a, false, b, (short)0, c,
                                                 false, false);
}

// ---- fragment layout helpers (contiguous k = 16*half + e in A and B) ----
static __device__ AI inline int a_kidx(int e, int half) {
  return half * 16 + e;
}

// A: 16x32 (MxK), row-major [M][K] bf16 source
static __device__ AI inline v16bf load_a_bf16(const bf16* A, int ld, int row0, int k0, int lane) {
  int m = lane & 15, half = lane >> 4;
  v16bf a;
#pragma unroll
  for (int e = 0; e < 16; ++e)
    a[e] = A[(size_t)(row0 + m) * ld + k0 + a_kidx(e, half)];
  return a;
}

// B: 32x16 (KxN), row-major [K][N] source
static __device__ AI inline v16bf load_b_bf16(const bf16* Bm, int ld, int k0, int col0, int lane) {
  int n = lane & 15, kh = lane >> 4;
  v16bf b;
#pragma unroll
  for (int e = 0; e < 16; ++e)
    b[e] = Bm[(size_t)(k0 + kh * 16 + e) * ld + col0 + n];
  return b;
}

// B from transposed storage: memory is [col][K] (contiguous K per lane)
static __device__ AI inline v16bf load_bT_bf16(const bf16* Bt, int ld, int col0, int k0, int lane) {
  int n = lane & 15, kh = lane >> 4;
  v16bf b;
#pragma unroll
  for (int e = 0; e < 16; ++e)
    b[e] = Bt[(size_t)(col0 + n) * ld + k0 + kh * 16 + e];
  return b;
}

static __device__ AI inline float half_reduce_max(float v) {
  v = fmaxf(v, __shfl_xor(v, 1, 32));
  v = fmaxf(v, __shfl_xor(v, 2, 32));
  v = fmaxf(v, __shfl_xor(v, 4, 32));
  v = fmaxf(v, __shfl_xor(v, 8, 32));
  return v;
}
static __device__ AI inline float half_reduce_sum(float v) {
  v += __shfl_xor(v, 1, 32);
  v += __shfl_xor(v, 2, 32);
  v += __shfl_xor(v, 4, 32);
  v += __shfl_xor(v, 8, 32);
  return v;
}

// wave-private LDS RAW/WAR hazard (cross-lane transpose)
static __device__ AI inline void wave_lds_sync() {
  __builtin_amdgcn_wave_barrier();
  asm volatile("s_wait_dscnt 0x0" ::: "memory");
  __builtin_amdgcn_wave_barrier();
}

// ---- CDNA5 async copy global -> LDS (ASYNCcnt) ----
static __device__ AI inline void async_copy_b128(unsigned lds_off, const void* gaddr) {
  asm volatile("global_load_async_to_lds_b128 %0, %1, off"
               :: "v"(lds_off), "v"(gaddr) : "memory");
}
template <int NN> static __device__ AI inline void wait_async_le() {
  asm volatile("s_wait_asynccnt %0" :: "n"(NN) : "memory");
}

#ifdef HAVE_TDM
// ---- Tensor Data Mover: 2D tile load, bf16 elements (ISA ch.8 D# layout) ----
static __device__ AI inline void tdm_load_2d_bf16(unsigned lds_addr, const void* gaddr,
                                                  unsigned tensor_d0, unsigned tensor_d1,
                                                  unsigned d0_stride,
                                                  unsigned tile_d0, unsigned tile_d1) {
  v4u g0;
  g0[0] = 1u;                                         // count=1, user mode
  g0[1] = lds_addr;                                   // lds_addr [63:32]
  g0[2] = (unsigned)(size_t)gaddr;                    // global_addr low
  g0[3] = ((unsigned)((size_t)gaddr >> 32) & 0x01FFFFFFu) | 0x80000000u;  // hi + type=2
  v8i g1;
  g1[0] = (1 << 16);                                  // data_size = 2 bytes
  g1[1] = (int)(tensor_d0 << 16);                     // tensor_dim0[15:0] @ [63:48]
  g1[2] = (int)((tensor_d0 >> 16) | (tensor_d1 << 16));   // d0 hi, d1 lo
  g1[3] = (int)((tensor_d1 >> 16) | (tile_d0 << 16));     // d1 hi, tile_dim0
  g1[4] = (int)tile_d1;                               // tile_dim1 (tile_dim2=0)
  g1[5] = (int)d0_stride;                             // tensor_dim0_stride low 32
  g1[6] = 0;
  g1[7] = 0;
  v4i z4 = {0, 0, 0, 0};
#if __clang_major__ >= 23
  v8i z8 = {0, 0, 0, 0, 0, 0, 0, 0};
  __builtin_amdgcn_tensor_load_to_lds(g0, g1, z4, z4, z8, 0);
#else
  __builtin_amdgcn_tensor_load_to_lds(g0, g1, z4, z4, 0);
#endif
}
#endif

// ---------------- setup kernels ----------------
__global__ __launch_bounds__(256) void cvt_cp_kernel(const float* __restrict__ cp,
                                                     bf16* __restrict__ out, int count) {
  int i = blockIdx.x * blockDim.x + threadIdx.x;
  if (i < count) out[i] = (bf16)cp[i];
}

// Pack fp32 [M][K] weight into WMMA A-fragment order (bf16):
// out[(((kc*(M/16))+t)*32 + lane)*16 + e] = W[(t*16 + m)*K + kc*32 + a_kidx(e,half)]
__global__ __launch_bounds__(256) void pack_a_kernel(const float* __restrict__ W,
                                                     bf16* __restrict__ out,
                                                     int M, int K) {
  int i = blockIdx.x * blockDim.x + threadIdx.x;
  if (i >= M * K) return;
  int e = i & 15;
  int l = (i >> 4) & 31;
  int rest = i >> 9;
  int nt = M >> 4;
  int t = rest % nt, kc = rest / nt;
  int m = l & 15, half = l >> 4;
  out[i] = (bf16)W[(size_t)(t * 16 + m) * K + kc * 32 + a_kidx(e, half)];
}

// packed A-fragment load: one aligned 32B contiguous read per lane
static __device__ AI inline v16bf load_a_packed(const bf16* Wp, int nt, int kc, int t, int lane) {
  return *(const v16bf*)(Wp + (((size_t)kc * nt + t) * 32 + lane) * 16);
}

// ---------------- kernel 1: theta/phi/g projections ----------------
static __device__ AI inline void stage_x_panel(unsigned lds, const char* gsrc) {
#pragma unroll
  for (int i = 0; i < 4; ++i) {
    unsigned u = threadIdx.x + 128 * i;  // 0..511
    unsigned r = u >> 4, seg = u & 15;   // row 0..31, 16B segment 0..15
    async_copy_b128(lds + r * 256 + seg * 16,
                    gsrc + (size_t)r * (N_ * 4) + seg * 16);
  }
}

template <int TAG>
static __device__ AI inline void proj_one(const char* __restrict__ xbytes,
                                          const bf16* __restrict__ Wp,  // packed
                                          bf16* __restrict__ O, int block_n,
                                          int wave, int lane,
                                          float (&xp)[2][32][64]) {
  const int half = lane >> 4, col = lane & 15, n = lane & 15, kh = lane >> 4;
  const unsigned lds0 = (unsigned)(uintptr_t)&xp[0][0][0];
  const unsigned lds1 = (unsigned)(uintptr_t)&xp[1][0][0];
  const char* gp = xbytes + (size_t)block_n * 4;

  v8f acc[8];
#pragma unroll
  for (int t = 0; t < 8; ++t) acc[t] = (v8f){0.f,0.f,0.f,0.f,0.f,0.f,0.f,0.f};

  stage_x_panel(lds0, gp);  // chunk 0
  for (int i = 0; i < CH_ / 32; ++i) {
    const int cur = i & 1;
    if (i + 1 < CH_ / 32) {
      stage_x_panel(cur ? lds0 : lds1, gp + (size_t)(i + 1) * 32 * (N_ * 4));
      wait_async_le<4>();
    } else {
      wait_async_le<0>();
    }
    __syncthreads();
    // batch the 16 DS loads, then convert (avoids per-pair dscnt stalls)
    float tf[16];
#pragma unroll
    for (int e = 0; e < 16; ++e) tf[e] = xp[cur][kh * 16 + e][wave * 16 + n];
    v16bf bfrag;
#pragma unroll
    for (int e = 0; e < 16; ++e) bfrag[e] = (bf16)tf[e];
#pragma unroll
    for (int t = 0; t < 8; ++t)
      acc[t] = wmma_bf16(load_a_packed(Wp, 8, i, t, lane), bfrag, acc[t]);
    __syncthreads();
  }
  const int n0 = block_n + wave * 16;
#pragma unroll
  for (int t = 0; t < 8; ++t)
#pragma unroll
    for (int j = 0; j < 8; ++j)
      O[(size_t)(n0 + col) * FD_ + t * 16 + j + 8 * half] = (bf16)acc[t][j];
}

__global__ __launch_bounds__(128) void proj_kernel(
    const float* __restrict__ x, const bf16* __restrict__ wp_theta,
    const bf16* __restrict__ wp_phi, const bf16* __restrict__ wp_g,
    bf16* __restrict__ theta, bf16* __restrict__ phi, bf16* __restrict__ g) {
  __shared__ float xp[2][32][64];  // 16 KB double-buffered x panel
  const int lane = threadIdx.x & 31, wave = threadIdx.x >> 5;
  const int nb = N_ / 64;
  const int b  = blockIdx.x / nb;
  const int block_n = (blockIdx.x % nb) * 64;
  const char* xb = (const char*)(x + (size_t)b * CH_ * N_);
  proj_one<0>(xb, wp_theta, theta + (size_t)b * N_ * FD_, block_n, wave, lane, xp);
  proj_one<1>(xb, wp_phi,   phi   + (size_t)b * N_ * FD_, block_n, wave, lane, xp);
  proj_one<2>(xb, wp_g,     g     + (size_t)b * N_ * FD_, block_n, wave, lane, xp);
}

// ---------------- kernel 2a: pool cross-attention (flash over P) ----------------
__global__ __launch_bounds__(128) void attn_pool_kernel(
    const bf16* __restrict__ theta, const bf16* __restrict__ cp,
    bf16* __restrict__ feat) {
  __shared__ bf16 cbuf[2][FD_][32];   // double-buffered cp tile (16 KB)
  __shared__ bf16 pbuf[4][16][32];    // per-wave prob transpose (4 KB)
  const int lane = threadIdx.x & 31, wave = threadIdx.x >> 5;
  const int nb = N_ / 64;
  const int b  = blockIdx.x / nb;
  const int n0 = (blockIdx.x % nb) * 64 + wave * 16;
  const int half = lane >> 4, col = lane & 15;
  const bf16* th = theta + (size_t)b * N_ * FD_;
  const char* cpb = (const char*)cp;

  const unsigned lds_c0 = (unsigned)(uintptr_t)&cbuf[0][0][0];
  const unsigned lds_c1 = (unsigned)(uintptr_t)&cbuf[1][0][0];

  v16bf aq[4];
#pragma unroll
  for (int kk = 0; kk < 4; ++kk) aq[kk] = load_a_bf16(th, FD_, n0, kk * 32, lane);

  v8f acc[8];
  float mrow[8], lrow[8];
#pragma unroll
  for (int t = 0; t < 8; ++t) acc[t] = (v8f){0.f,0.f,0.f,0.f,0.f,0.f,0.f,0.f};
#pragma unroll
  for (int j = 0; j < 8; ++j) { mrow[j] = -1e30f; lrow[j] = 0.f; }

#ifdef HAVE_TDM
  if (wave == 0)
    tdm_load_2d_bf16(lds_c0, cpb, P_, FD_, P_, 32, FD_);
#else
#pragma unroll
  for (int i = 0; i < 4; ++i) {
    unsigned u = threadIdx.x + 128 * i, r = u >> 2, seg = u & 3;
    async_copy_b128(lds_c0 + r * 64 + seg * 16,
                    cpb + (size_t)r * (P_ * 2) + seg * 16);
  }
#endif

  for (int i = 0; i < P_ / 32; ++i) {
    const int cur = i & 1;
    const int p0n = (i + 1) * 32;
#ifdef HAVE_TDM
    if (wave == 0) {
      if (i + 1 < P_ / 32) {
        tdm_load_2d_bf16(cur ? lds_c0 : lds_c1, cpb + (size_t)p0n * 2,
                         P_, FD_, P_, 32, FD_);
        __builtin_amdgcn_s_wait_tensorcnt((short)1);
      } else {
        __builtin_amdgcn_s_wait_tensorcnt((short)0);
      }
    }
#else
    if (i + 1 < P_ / 32) {
      unsigned dst = cur ? lds_c0 : lds_c1;
#pragma unroll
      for (int q = 0; q < 4; ++q) {
        unsigned u = threadIdx.x + 128 * q, r = u >> 2, seg = u & 3;
        async_copy_b128(dst + r * 64 + seg * 16,
                        cpb + (size_t)r * (P_ * 2) + (size_t)p0n * 2 + seg * 16);
      }
      wait_async_le<4>();
    } else {
      wait_async_le<0>();
    }
#endif
    __syncthreads();
    const bf16* cb = &cbuf[cur][0][0];

    v8f S0 = (v8f){0.f,0.f,0.f,0.f,0.f,0.f,0.f,0.f};
    v8f S1 = S0;
#pragma unroll
    for (int kk = 0; kk < 4; ++kk) {
      S0 = wmma_bf16(aq[kk], load_b_bf16(cb, 32, kk * 32, 0,  lane), S0);
      S1 = wmma_bf16(aq[kk], load_b_bf16(cb, 32, kk * 32, 16, lane), S1);
    }
#pragma unroll
    for (int j = 0; j < 8; ++j) {
      float rm   = half_reduce_max(fmaxf(S0[j], S1[j]));
      float mnew = fmaxf(mrow[j], rm);
      float sc   = __expf(mrow[j] - mnew);
      float e0   = __expf(S0[j] - mnew);
      float e1   = __expf(S1[j] - mnew);
      lrow[j] = lrow[j] * sc + half_reduce_sum(e0 + e1);
      mrow[j] = mnew;
#pragma unroll
      for (int t = 0; t < 8; ++t) acc[t][j] *= sc;
      pbuf[wave][j + 8 * half][col]      = (bf16)e0;
      pbuf[wave][j + 8 * half][16 + col] = (bf16)e1;
    }
    wave_lds_sync();
    v16bf ap = load_a_bf16(&pbuf[wave][0][0], 32, 0, 0, lane);
    wave_lds_sync();
#pragma unroll
    for (int t = 0; t < 8; ++t)
      acc[t] = wmma_bf16(ap, load_bT_bf16(cb, 32, t * 16, 0, lane), acc[t]);
    __syncthreads();
  }
#pragma unroll
  for (int t = 0; t < 8; ++t)
#pragma unroll
    for (int j = 0; j < 8; ++j)
      feat[((size_t)b * N_ + n0 + j + 8 * half) * FD_ + t * 16 + col] =
          (bf16)(acc[t][j] / lrow[j]);
}

// ---------------- kernel 2b: self-attention (flash over N), feat += sa_out ----------------
__global__ __launch_bounds__(128) void attn_sa_kernel(
    const bf16* __restrict__ theta, const bf16* __restrict__ phi,
    const bf16* __restrict__ g, bf16* __restrict__ feat) {
  __shared__ bf16 kbuf[2][32][FD_];   // 16 KB
  __shared__ bf16 vbuf[2][32][FD_];   // 16 KB
  __shared__ bf16 pbuf[4][16][32];    // 4 KB
  const int lane = threadIdx.x & 31, wave = threadIdx.x >> 5;
  const int nb = N_ / 64;
  const int b  = blockIdx.x / nb;
  const int n0 = (blockIdx.x % nb) * 64 + wave * 16;
  const int half = lane >> 4, col = lane & 15;
  const bf16* th = theta + (size_t)b * N_ * FD_;
  const char* phb = (const char*)(phi + (size_t)b * N_ * FD_);
  const char* gvb = (const char*)(g   + (size_t)b * N_ * FD_);

  unsigned lds_k[2] = {(unsigned)(uintptr_t)&kbuf[0][0][0],
                       (unsigned)(uintptr_t)&kbuf[1][0][0]};
  unsigned lds_v[2] = {(unsigned)(uintptr_t)&vbuf[0][0][0],
                       (unsigned)(uintptr_t)&vbuf[1][0][0]};

  v16bf aq[4];
#pragma unroll
  for (int kk = 0; kk < 4; ++kk) aq[kk] = load_a_bf16(th, FD_, n0, kk * 32, lane);

  v8f acc[8];
  float mrow[8], lrow[8];
#pragma unroll
  for (int t = 0; t < 8; ++t) acc[t] = (v8f){0.f,0.f,0.f,0.f,0.f,0.f,0.f,0.f};
#pragma unroll
  for (int j = 0; j < 8; ++j) { mrow[j] = -1e30f; lrow[j] = 0.f; }

#pragma unroll
  for (int q = 0; q < 4; ++q) {
    unsigned o = (threadIdx.x + 128 * q) * 16;
    async_copy_b128(lds_k[0] + o, phb + o);
    async_copy_b128(lds_v[0] + o, gvb + o);
  }

  for (int i = 0; i < N_ / 32; ++i) {
    const int cur = i & 1;
    if (i + 1 < N_ / 32) {
      const size_t nxt = (size_t)(i + 1) * 32 * FD_ * 2;
#pragma unroll
      for (int q = 0; q < 4; ++q) {
        unsigned o = (threadIdx.x + 128 * q) * 16;
        async_copy_b128(lds_k[cur ^ 1] + o, phb + nxt + o);
        async_copy_b128(lds_v[cur ^ 1] + o, gvb + nxt + o);
      }
      wait_async_le<8>();
    } else {
      wait_async_le<0>();
    }
    __syncthreads();
    const bf16* kb = &kbuf[cur][0][0];
    const bf16* vb = &vbuf[cur][0][0];

    v8f S0 = (v8f){0.f,0.f,0.f,0.f,0.f,0.f,0.f,0.f};
    v8f S1 = S0;
#pragma unroll
    for (int kk = 0; kk < 4; ++kk) {
      S0 = wmma_bf16(aq[kk], load_bT_bf16(kb, FD_, 0,  kk * 32, lane), S0);
      S1 = wmma_bf16(aq[kk], load_bT_bf16(kb, FD_, 16, kk * 32, lane), S1);
    }
#pragma unroll
    for (int j = 0; j < 8; ++j) {
      float rm   = half_reduce_max(fmaxf(S0[j], S1[j]));
      float mnew = fmaxf(mrow[j], rm);
      float sc   = __expf(mrow[j] - mnew);
      float e0   = __expf(S0[j] - mnew);
      float e1   = __expf(S1[j] - mnew);
      lrow[j] = lrow[j] * sc + half_reduce_sum(e0 + e1);
      mrow[j] = mnew;
#pragma unroll
      for (int t = 0; t < 8; ++t) acc[t][j] *= sc;
      pbuf[wave][j + 8 * half][col]      = (bf16)e0;
      pbuf[wave][j + 8 * half][16 + col] = (bf16)e1;
    }
    wave_lds_sync();
    v16bf ap = load_a_bf16(&pbuf[wave][0][0], 32, 0, 0, lane);
    wave_lds_sync();
#pragma unroll
    for (int t = 0; t < 8; ++t)
      acc[t] = wmma_bf16(ap, load_b_bf16(vb, FD_, 0, t * 16, lane), acc[t]);
    __syncthreads();
  }
#pragma unroll
  for (int t = 0; t < 8; ++t)
#pragma unroll
    for (int j = 0; j < 8; ++j) {
      size_t idx = ((size_t)b * N_ + n0 + j + 8 * half) * FD_ + t * 16 + col;
      feat[idx] = (bf16)((float)feat[idx] + acc[t][j] / lrow[j]);
    }
}

// ---------------- kernel 3: output projection + gamma residual ----------------
__global__ __launch_bounds__(128) void out_kernel(
    const float* __restrict__ x, const bf16* __restrict__ wp_o,  // packed
    const float* __restrict__ gamma, const bf16* __restrict__ feat,
    float* __restrict__ out) {
  const int lane = threadIdx.x & 31, wave = threadIdx.x >> 5;
  const int nb = N_ / 64, cb = CH_ / 16;
  int bid = blockIdx.x;
  const int b   = bid / (cb * nb);
  const int rem = bid % (cb * nb);
  const int ct  = rem / nb;           // c-tile index
  const int n0  = (rem % nb) * 64 + wave * 16;
  const int half = lane >> 4, col = lane & 15;
  const bf16* fb = feat + (size_t)b * N_ * FD_;

  v8f acc = (v8f){0.f,0.f,0.f,0.f,0.f,0.f,0.f,0.f};
#pragma unroll
  for (int kk = 0; kk < 4; ++kk) {
    v16bf a  = load_a_packed(wp_o, CH_ / 16, kk, ct, lane);
    v16bf bb = load_bT_bf16(fb, FD_, n0, kk * 32, lane);  // B[k=f][n]=feat[n][f]
    acc = wmma_bf16(a, bb, acc);
  }
  const float gm = gamma[0];
#pragma unroll
  for (int j = 0; j < 8; ++j) {
    int c = ct * 16 + j + 8 * half;
    int n = n0 + col;
    size_t idx = ((size_t)b * CH_ + c) * N_ + n;
    out[idx] = gm * acc[j] + x[idx];
  }
}

extern "C" void kernel_launch(void* const* d_in, const int* in_sizes, int n_in,
                              void* d_out, int out_size, void* d_ws, size_t ws_size,
                              hipStream_t stream) {
  const float* x        = (const float*)d_in[0];
  const float* w_theta  = (const float*)d_in[1];
  const float* w_phi    = (const float*)d_in[2];
  const float* w_g      = (const float*)d_in[3];
  const float* w_o      = (const float*)d_in[4];
  const float* gamma    = (const float*)d_in[5];
  const float* cp_f32   = (const float*)d_in[6];
  float* out = (float*)d_out;

  const size_t SZ  = (size_t)B_ * N_ * FD_ * sizeof(bf16);   // 8 MB each
  const size_t WSZ = (size_t)FD_ * CH_ * sizeof(bf16);       // 256 KB each
  char* ws = (char*)d_ws;
  bf16* theta = (bf16*)(ws + 0 * SZ);
  bf16* phi   = (bf16*)(ws + 1 * SZ);
  bf16* g     = (bf16*)(ws + 2 * SZ);
  bf16* feat  = (bf16*)(ws + 3 * SZ);
  bf16* cp    = (bf16*)(ws + 4 * SZ);                        // 512 KB
  bf16* wp_th = (bf16*)(ws + 4 * SZ + 1 * 524288);
  bf16* wp_ph = (bf16*)(ws + 4 * SZ + 1 * 524288 + 1 * WSZ);
  bf16* wp_g  = (bf16*)(ws + 4 * SZ + 1 * 524288 + 2 * WSZ);
  bf16* wp_o  = (bf16*)(ws + 4 * SZ + 1 * 524288 + 3 * WSZ);

  cvt_cp_kernel<<<(FD_ * P_ + 255) / 256, 256, 0, stream>>>(cp_f32, cp, FD_ * P_);
  pack_a_kernel<<<(FD_ * CH_ + 255) / 256, 256, 0, stream>>>(w_theta, wp_th, FD_, CH_);
  pack_a_kernel<<<(FD_ * CH_ + 255) / 256, 256, 0, stream>>>(w_phi,   wp_ph, FD_, CH_);
  pack_a_kernel<<<(FD_ * CH_ + 255) / 256, 256, 0, stream>>>(w_g,     wp_g,  FD_, CH_);
  pack_a_kernel<<<(CH_ * FD_ + 255) / 256, 256, 0, stream>>>(w_o,     wp_o,  CH_, FD_);

  proj_kernel<<<B_ * (N_ / 64), 128, 0, stream>>>(x, wp_th, wp_ph, wp_g, theta, phi, g);
  attn_pool_kernel<<<B_ * (N_ / 64), 128, 0, stream>>>(theta, cp, feat);
  attn_sa_kernel<<<B_ * (N_ / 64), 128, 0, stream>>>(theta, phi, g, feat);
  out_kernel<<<B_ * (CH_ / 16) * (N_ / 64), 128, 0, stream>>>(x, wp_o, gamma, feat, out);
}
